// FNCModel_68006512165157
// MI455X (gfx1250) — matrix operand order repaired and checked by reading, not compiled, not run on hardware
//
#include <hip/hip_runtime.h>
#include <hip/hip_bf16.h>

// ---------------------------------------------------------------------------
// Types / helpers
// ---------------------------------------------------------------------------
typedef __bf16 v16bf __attribute__((ext_vector_type(16)));
typedef float  v8f   __attribute__((ext_vector_type(8)));

union FragAB {
    uint4 q[2];
    v16bf v;
};

__device__ __forceinline__ unsigned short f2bf(float f) {
    // round-to-nearest-even fp32 -> bf16
    unsigned u = __float_as_uint(f);
    unsigned r = u + 0x7FFFu + ((u >> 16) & 1u);
    return (unsigned short)(r >> 16);
}
__device__ __forceinline__ float bf2f(unsigned short h) {
    return __uint_as_float(((unsigned)h) << 16);
}

// Model constants
static constexpr int B    = 128;
static constexpr int E    = 300;
static constexpr int KX   = 320;   // padded x width (multiple of 32)
static constexpr int H    = 256;
static constexpr int TH_  = 32;
static constexpr int TB_  = 512;

// ---------------------------------------------------------------------------
// Kernel 1: embedding gather -> bf16 X[t][b][320] (cols 300..319 zero)
// ---------------------------------------------------------------------------
__global__ void embed_kernel(const int* __restrict__ ids,
                             const float* __restrict__ emb,
                             unsigned short* __restrict__ X, int T) {
    long long idx = (long long)blockIdx.x * blockDim.x + threadIdx.x;
    long long total = (long long)T * B * KX;
    if (idx >= total) return;
    int kk = (int)(idx % KX);
    int rb = (int)(idx / KX);     // t*B + b
    int b  = rb % B;
    int t  = rb / B;
    int id = ids[b * T + t];      // NF == 1
    float v = (kk < E) ? emb[(long long)id * E + kk] : 0.0f;
    X[idx] = f2bf(v);
}

// ---------------------------------------------------------------------------
// Kernel 2: seqlen = count of nonzero ids (pad embedding row is zero)
// ---------------------------------------------------------------------------
__global__ void seqlen_kernel(const int* __restrict__ h_ids,
                              const int* __restrict__ b_ids,
                              int* __restrict__ seqlen) {
    int s = blockIdx.x;           // 0 = headline, 1 = body
    int b = threadIdx.x;
    const int* ids = (s == 0) ? h_ids : b_ids;
    int T = (s == 0) ? TH_ : TB_;
    int c = 0;
    for (int t = 0; t < T; ++t) c += (ids[b * T + t] != 0);
    seqlen[s * B + b] = c;
}

// ---------------------------------------------------------------------------
// Kernel 3: pack fp32 weights into WMMA B-fragment tiles, bf16.
// dst layout: idx = ((nt*KT + kt)*32 + lane)*16 + j
//   K = kt*32 + (j>>3)*16 + (lane>>4)*8 + (j&7) , N = nt*16 + (lane&15)
// l0: K<300 -> row K ; 300<=K<320 -> zero pad ; K>=320 -> row K-20 (h part)
// ---------------------------------------------------------------------------
__global__ void pack_w_kernel(const float* __restrict__ src,
                              unsigned short* __restrict__ dst,
                              int N, int KT, int Ksrc, int l0) {
    int idx = blockIdx.x * blockDim.x + threadIdx.x;
    int total = (N / 16) * KT * 512;
    if (idx >= total) return;
    int j    = idx & 15;
    int lane = (idx >> 4) & 31;
    int rest = idx >> 9;
    int kt   = rest % KT;
    int nt   = rest / KT;
    int K = kt * 32 + (j >> 3) * 16 + (lane >> 4) * 8 + (j & 7);
    int n = nt * 16 + (lane & 15);
    int row;
    if (l0) row = (K < E) ? K : ((K < KX) ? -1 : K - (KX - E));
    else    row = K;
    float v = (row >= 0 && row < Ksrc) ? src[(long long)row * N + n] : 0.0f;
    dst[idx] = f2bf(v);
}

// ---------------------------------------------------------------------------
// Kernel 4: persistent fused 2-layer GRU recurrence. 2 blocks (streams),
// 512 threads = 16 wave32. LDS: h0,h1,RH,U bf16 [128][256] + seqlen = ~256 KB.
// ---------------------------------------------------------------------------
struct GruArgs {
    const unsigned short* Xh;           // [32][128][320]
    const unsigned short* Xb;           // [512][128][320]
    const unsigned short* Wg0[2];       // packed [32][18][32][16]
    const unsigned short* Wc0[2];       // packed [16][18][32][16]
    const unsigned short* Wg1[2];       // packed [32][16][32][16]
    const unsigned short* Wc1[2];       // packed [16][16][32][16]
    const float* bg0[2]; const float* bc0[2];
    const float* bg1[2]; const float* bc1[2];
    const int*   seqlen;                // [2][128]
    unsigned short* out0;               // [2][128][256] bf16 (L0->L1 staging)
    float*       hfinal;                // [2][128][256] fp32
};

// One GRU layer step. L0 selects compile-time K geometry and x source stride,
// so both k-loops are branch-free and fully unrolled (loads clause-batched,
// WMMAs back-to-back).
template<bool L0>
__device__ __forceinline__ void gru_layer_step(
    const unsigned short* __restrict__ xbase,   // x rows: xbase + m*XS
    const unsigned short* __restrict__ Wg,
    const unsigned short* __restrict__ Wc,
    const float* __restrict__ bg,
    const float* __restrict__ bc,
    unsigned short* hb, unsigned short* RH, unsigned short* U,
    const unsigned short* slLDS,
    unsigned short* OUT0,
    int t, int wv, int ln, int hi, int lm)
{
    constexpr int KT  = L0 ? 18 : 16;   // total K tiles (576 / 512)
    constexpr int KHT = KT - (L0 ? 10 : 8);  // h-part K tiles (8)
    constexpr int KXT = KT - KHT;       // x-part K tiles (10 / 8)
    constexpr int XS  = L0 ? KX : H;    // x row stride (halfs)

    // ---------------- phase G: gates = sigmoid([x|h] @ Wg + bg) ------------
    for (int tt = wv; tt < 256; tt += 16) {
        const int mt = tt & 7, nt = tt >> 3;
        const int m  = mt * 16 + lm;
        const unsigned short* ax = xbase + (size_t)m * XS + hi * 8;
        const unsigned short* ah = hb + (size_t)m * H + hi * 8;
        const unsigned short* bw = Wg + (size_t)nt * KT * 512 + ln * 16;
        __builtin_prefetch(bw + 2 * KT * 512, 0, 3);   // next owned tile (WGP)
        v8f acc = {};
#pragma unroll
        for (int kt = 0; kt < KXT; ++kt) {
            FragAB a, b;
            a.q[0] = *(const uint4*)(ax + kt * 32);
            a.q[1] = *(const uint4*)(ax + kt * 32 + 16);
            b.q[0] = *(const uint4*)(bw + kt * 512);
            b.q[1] = *(const uint4*)(bw + kt * 512 + 8);
            acc = __builtin_amdgcn_wmma_f32_16x16x32_bf16(
                false, a.v, false, b.v, (short)0, acc, false, false);
        }
#pragma unroll
        for (int kt = 0; kt < KHT; ++kt) {
            FragAB a, b;
            a.q[0] = *(const uint4*)(ah + kt * 32);
            a.q[1] = *(const uint4*)(ah + kt * 32 + 16);
            b.q[0] = *(const uint4*)(bw + (KXT + kt) * 512);
            b.q[1] = *(const uint4*)(bw + (KXT + kt) * 512 + 8);
            acc = __builtin_amdgcn_wmma_f32_16x16x32_bf16(
                false, a.v, false, b.v, (short)0, acc, false, false);
        }
        if (nt < 16) {            // reset gate -> r * h (uniform branch)
#pragma unroll
            for (int v = 0; v < 8; ++v) {
                const int mm = mt * 16 + v + hi * 8;
                const int nn = nt * 16 + lm;
                float g = acc[v] + bg[nn];
                g = 1.0f / (1.0f + __expf(-g));
                RH[mm * H + nn] = f2bf(g * bf2f(hb[mm * H + nn]));
            }
        } else {                  // update gate
#pragma unroll
            for (int v = 0; v < 8; ++v) {
                const int mm = mt * 16 + v + hi * 8;
                const int nn = (nt - 16) * 16 + lm;
                float g = acc[v] + bg[nn + H];
                g = 1.0f / (1.0f + __expf(-g));
                U[mm * H + nn] = f2bf(g);
            }
        }
    }
    __syncthreads();

    // ---------- phase C: c = tanh([x|r*h] @ Wc + bc), fused h update -------
    // matmul reads X/OUT0 + RH only; epilogue writes hb (+OUT0 for L0) only:
    // no cross-wave overlap, so no mid-phase barrier needed.
    for (int tt = wv; tt < 128; tt += 16) {
        const int mt = tt & 7, nt = tt >> 3;
        const int m  = mt * 16 + lm;
        const unsigned short* ax = xbase + (size_t)m * XS + hi * 8;
        const unsigned short* ar = RH + (size_t)m * H + hi * 8;
        const unsigned short* bw = Wc + (size_t)nt * KT * 512 + ln * 16;
        __builtin_prefetch(bw + 2 * KT * 512, 0, 3);
        v8f acc = {};
#pragma unroll
        for (int kt = 0; kt < KXT; ++kt) {
            FragAB a, b;
            a.q[0] = *(const uint4*)(ax + kt * 32);
            a.q[1] = *(const uint4*)(ax + kt * 32 + 16);
            b.q[0] = *(const uint4*)(bw + kt * 512);
            b.q[1] = *(const uint4*)(bw + kt * 512 + 8);
            acc = __builtin_amdgcn_wmma_f32_16x16x32_bf16(
                false, a.v, false, b.v, (short)0, acc, false, false);
        }
#pragma unroll
        for (int kt = 0; kt < KHT; ++kt) {
            FragAB a, b;
            a.q[0] = *(const uint4*)(ar + kt * 32);
            a.q[1] = *(const uint4*)(ar + kt * 32 + 16);
            b.q[0] = *(const uint4*)(bw + (KXT + kt) * 512);
            b.q[1] = *(const uint4*)(bw + (KXT + kt) * 512 + 8);
            acc = __builtin_amdgcn_wmma_f32_16x16x32_bf16(
                false, a.v, false, b.v, (short)0, acc, false, false);
        }
#pragma unroll
        for (int v = 0; v < 8; ++v) {
            const int mm = mt * 16 + v + hi * 8;
            const int nn = nt * 16 + lm;
            float c  = tanhf(acc[v] + bc[nn]);
            float u  = bf2f(U[mm * H + nn]);
            float h  = bf2f(hb[mm * H + nn]);
            float nh = u * h + (1.0f - u) * c;
            bool valid = t < (int)slLDS[mm];
            hb[mm * H + nn] = f2bf(valid ? nh : h);
            if (L0) OUT0[mm * H + nn] = f2bf(valid ? nh : 0.0f);
        }
    }
    __threadfence_block();   // OUT0 (global) visible to whole block
    __syncthreads();
}

__global__ __launch_bounds__(512)
void gru_kernel(GruArgs ga) {
    const int s   = blockIdx.x;
    const int tid = threadIdx.x;
    const int wv  = tid >> 5;       // wave 0..15
    const int ln  = tid & 31;
    const int hi  = ln >> 4;        // lane half
    const int lm  = ln & 15;
    const int T   = (s == 0) ? TH_ : TB_;
    const unsigned short* Xs = (s == 0) ? ga.Xh : ga.Xb;
    unsigned short* OUT0 = ga.out0 + (size_t)s * B * H;

    extern __shared__ unsigned short sm[];
    unsigned short* hbf0  = sm;              // 128*256
    unsigned short* hbf1  = sm + 32768;
    unsigned short* RH    = sm + 65536;      // r (*) h
    unsigned short* U     = sm + 98304;      // update gate
    unsigned short* slLDS = sm + 131072;     // seqlen[128] (u16)

    for (int i = tid; i < 65536; i += 512) sm[i] = 0;   // h0 = h1 = 0
    if (tid < B) slLDS[tid] = (unsigned short)ga.seqlen[s * B + tid];
    __syncthreads();

    const unsigned short* Wg0 = ga.Wg0[s];
    const unsigned short* Wc0 = ga.Wc0[s];
    const unsigned short* Wg1 = ga.Wg1[s];
    const unsigned short* Wc1 = ga.Wc1[s];
    const float* bg0 = ga.bg0[s]; const float* bc0 = ga.bc0[s];
    const float* bg1 = ga.bg1[s]; const float* bc1 = ga.bc1[s];

    for (int t = 0; t < T; ++t) {
        gru_layer_step<true>(Xs + (size_t)t * B * KX, Wg0, Wc0, bg0, bc0,
                             hbf0, RH, U, slLDS, OUT0,
                             t, wv, ln, hi, lm);
        gru_layer_step<false>(OUT0, Wg1, Wc1, bg1, bc1,
                              hbf1, RH, U, slLDS, OUT0,
                              t, wv, ln, hi, lm);
    }

    // final state = layer-1 h
    for (int i = tid; i < B * H; i += 512)
        ga.hfinal[(size_t)s * B * H + i] = bf2f(hbf1[i]);
}

// ---------------------------------------------------------------------------
// Kernel 5: out[b][c] = concat(h_head, h_body)[b,:] @ W_pred + b_pred
// ---------------------------------------------------------------------------
__global__ void pred_kernel(const float* __restrict__ hfin,
                            const float* __restrict__ Wp,
                            const float* __restrict__ bp,
                            float* __restrict__ out) {
    int idx = blockIdx.x * blockDim.x + threadIdx.x;   // 128*4
    if (idx >= B * 4) return;
    int b = idx >> 2, c = idx & 3;
    const float* hh = hfin + (size_t)b * H;                 // head
    const float* hb = hfin + (size_t)B * H + (size_t)b * H; // body
    float acc = bp[c];
    for (int k = 0; k < H; ++k) acc += hh[k] * Wp[k * 4 + c];
    for (int k = 0; k < H; ++k) acc += hb[k] * Wp[(H + k) * 4 + c];
    out[idx] = acc;
}

// ---------------------------------------------------------------------------
// Host launcher
// ---------------------------------------------------------------------------
extern "C" void kernel_launch(void* const* d_in, const int* in_sizes, int n_in,
                              void* d_out, int out_size, void* d_ws, size_t ws_size,
                              hipStream_t stream) {
    const int*   h_ids  = (const int*)  d_in[0];
    const int*   b_ids  = (const int*)  d_in[1];
    const float* emb    = (const float*)d_in[2];
    const float* W_pred = (const float*)d_in[3];
    const float* b_pred = (const float*)d_in[4];
    // hd0:5..8  hd1:9..12  bd0:13..16  bd1:17..20  (Wg, bg, Wc, bc)
    const float* Wg0f[2] = { (const float*)d_in[5],  (const float*)d_in[13] };
    const float* bg0f[2] = { (const float*)d_in[6],  (const float*)d_in[14] };
    const float* Wc0f[2] = { (const float*)d_in[7],  (const float*)d_in[15] };
    const float* bc0f[2] = { (const float*)d_in[8],  (const float*)d_in[16] };
    const float* Wg1f[2] = { (const float*)d_in[9],  (const float*)d_in[17] };
    const float* bg1f[2] = { (const float*)d_in[10], (const float*)d_in[18] };
    const float* Wc1f[2] = { (const float*)d_in[11], (const float*)d_in[19] };
    const float* bc1f[2] = { (const float*)d_in[12], (const float*)d_in[20] };

    char* wsb = (char*)d_ws;
    size_t off = 0;
    auto alloc = [&](size_t bytes) -> char* {
        off = (off + 255) & ~(size_t)255;
        char* p = wsb + off;
        off += bytes;
        return p;
    };

    unsigned short* Xh = (unsigned short*)alloc((size_t)TH_ * B * KX * 2);
    unsigned short* Xb = (unsigned short*)alloc((size_t)TB_ * B * KX * 2);
    unsigned short* Wg0p[2], *Wc0p[2], *Wg1p[2], *Wc1p[2];
    for (int s = 0; s < 2; ++s) {
        Wg0p[s] = (unsigned short*)alloc((size_t)32 * 18 * 512 * 2);
        Wc0p[s] = (unsigned short*)alloc((size_t)16 * 18 * 512 * 2);
        Wg1p[s] = (unsigned short*)alloc((size_t)32 * 16 * 512 * 2);
        Wc1p[s] = (unsigned short*)alloc((size_t)16 * 16 * 512 * 2);
    }
    int*            seqlen = (int*)           alloc(2 * B * sizeof(int));
    unsigned short* out0   = (unsigned short*)alloc((size_t)2 * B * H * 2);
    float*          hfinal = (float*)         alloc((size_t)2 * B * H * sizeof(float));
    (void)ws_size; (void)n_in; (void)in_sizes; (void)out_size;

    // 1) embedding gather (bf16, padded to 320)
    {
        long long tot_h = (long long)TH_ * B * KX;
        long long tot_b = (long long)TB_ * B * KX;
        embed_kernel<<<(unsigned)((tot_h + 255) / 256), 256, 0, stream>>>(h_ids, emb, Xh, TH_);
        embed_kernel<<<(unsigned)((tot_b + 255) / 256), 256, 0, stream>>>(b_ids, emb, Xb, TB_);
    }
    // 2) sequence lengths
    seqlen_kernel<<<2, 128, 0, stream>>>(h_ids, b_ids, seqlen);
    // 3) weight packing into WMMA B-fragment tiles
    for (int s = 0; s < 2; ++s) {
        int tot;
        tot = 32 * 18 * 512;
        pack_w_kernel<<<(tot + 255) / 256, 256, 0, stream>>>(Wg0f[s], Wg0p[s], 512, 18, E + H, 1);
        tot = 16 * 18 * 512;
        pack_w_kernel<<<(tot + 255) / 256, 256, 0, stream>>>(Wc0f[s], Wc0p[s], 256, 18, E + H, 1);
        tot = 32 * 16 * 512;
        pack_w_kernel<<<(tot + 255) / 256, 256, 0, stream>>>(Wg1f[s], Wg1p[s], 512, 16, 2 * H, 0);
        tot = 16 * 16 * 512;
        pack_w_kernel<<<(tot + 255) / 256, 256, 0, stream>>>(Wc1f[s], Wc1p[s], 256, 16, 2 * H, 0);
    }
    // 4) fused persistent GRU recurrence (2 streams, ~256 KB LDS each)
    GruArgs ga;
    ga.Xh = Xh; ga.Xb = Xb;
    for (int s = 0; s < 2; ++s) {
        ga.Wg0[s] = Wg0p[s]; ga.Wc0[s] = Wc0p[s];
        ga.Wg1[s] = Wg1p[s]; ga.Wc1[s] = Wc1p[s];
        ga.bg0[s] = bg0f[s]; ga.bc0[s] = bc0f[s];
        ga.bg1[s] = bg1f[s]; ga.bc1[s] = bc1f[s];
    }
    ga.seqlen = seqlen; ga.out0 = out0; ga.hfinal = hfinal;
    gru_kernel<<<2, 512, 262400, stream>>>(ga);
    // 5) classifier head
    pred_kernel<<<2, 256, 0, stream>>>(hfinal, W_pred, b_pred, (float*)d_out);
}
